// HSTU_52536039965138
// MI455X (gfx1250) — matrix operand order repaired and checked by reading, not compiled, not run on hardware
//
#include <hip/hip_runtime.h>
#include <hip/hip_bf16.h>

// ---------------- types & helpers ----------------
typedef __attribute__((ext_vector_type(16))) __bf16 v16bf;
typedef __attribute__((ext_vector_type(8)))  __bf16 v8bf;
typedef __attribute__((ext_vector_type(4)))  __bf16 v4bf;
typedef __attribute__((ext_vector_type(8)))  float  v8f;
typedef __attribute__((ext_vector_type(2)))  unsigned v2u;
typedef __attribute__((ext_vector_type(4)))  unsigned v4u;

#define BQ   4
#define LQ   1024
#define BL   4096          // B*L
#define DIM  512
#define D4   2048
#define NH   8
#define HD   64
#define NITEMS 20001
#define NPAD   20032       // 313*64

__device__ __forceinline__ unsigned short bfbits(float f) {
  unsigned u = __builtin_bit_cast(unsigned, f);
  return (unsigned short)((u + 0x7FFFu + ((u >> 16) & 1u)) >> 16);  // RNE
}
__device__ __forceinline__ __bf16 f2bf(float f) {
  unsigned short h = bfbits(f);
  return __builtin_bit_cast(__bf16, h);
}
__device__ __forceinline__ float bf2f(__bf16 h) {
  unsigned u = ((unsigned)__builtin_bit_cast(unsigned short, h)) << 16;
  return __builtin_bit_cast(float, u);
}
// two f32 -> packed bf16 (RNE): 2x round-add + one v_perm_b32 merge
__device__ __forceinline__ unsigned pack2(float a, float b) {
  unsigned ua = __builtin_bit_cast(unsigned, a);
  unsigned ub = __builtin_bit_cast(unsigned, b);
  unsigned ra = ua + 0x7FFFu + ((ua >> 16) & 1u);
  unsigned rb = ub + 0x7FFFu + ((ub >> 16) & 1u);
  return __builtin_amdgcn_perm(rb, ra, 0x07060302);  // [ra.hi16 | rb.hi16]
}
// fast SiLU: v * v_rcp_f32(1+exp(-v))
__device__ __forceinline__ float silu(float v) {
  return v * __builtin_amdgcn_rcpf(1.0f + __expf(-v));
}

__device__ __forceinline__ v8f wmma_bf16(v16bf a, v16bf b, v8f c) {
  return __builtin_amdgcn_wmma_f32_16x16x32_bf16(false, a, false, b, (short)0, c,
                                                 false, false);
}

// Async 16B copy global->LDS, tracked by ASYNCcnt (no VGPR round trip).
__device__ __forceinline__ void async_b128(unsigned lds_off, const void* base,
                                           unsigned voff) {
  asm volatile("global_load_async_to_lds_b128 %0, %1, %2"
               :: "v"(lds_off), "v"(voff), "s"(base) : "memory");
}
__device__ __forceinline__ void wait_async0() {
  asm volatile("s_wait_asynccnt 0x0" ::: "memory");
}
__device__ __forceinline__ void wait_async6() {
  asm volatile("s_wait_asynccnt 0x6" ::: "memory");
}
__device__ __forceinline__ unsigned lds_addr(const void* p) {
  return (unsigned)(uintptr_t)p;  // low 32 bits of generic LDS pointer
}

// A fragment from row-major [16][>=32] bf16 tile (row stride in elems).
__device__ __forceinline__ v16bf load_a_frag(const __bf16* T, int stride, int lane) {
  const int row = lane & 15, hi = lane >> 4;
  const __bf16* p = T + row * stride + hi * 8;
  v8bf lo = *(const v8bf*)p;
  v8bf hh = *(const v8bf*)(p + 16);
  return __builtin_shufflevector(lo, hh, 0, 1, 2, 3, 4, 5, 6, 7,
                                 8, 9, 10, 11, 12, 13, 14, 15);
}

// B fragment from N-major [n][>=32 k] tile.
__device__ __forceinline__ v16bf load_b_frag(const __bf16* T, int stride, int col0,
                                             int lane) {
  const __bf16* p = T + (col0 + (lane & 15)) * stride + (lane >> 4) * 8;
  v8bf lo = *(const v8bf*)p;
  v8bf hh = *(const v8bf*)(p + 16);
  return __builtin_shufflevector(lo, hh, 0, 1, 2, 3, 4, 5, 6, 7,
                                 8, 9, 10, 11, 12, 13, 14, 15);
}

__device__ __forceinline__ float block_reduce(float v, float* red, int tid) {
  red[tid] = v;
  __syncthreads();
  for (int off = 64; off > 0; off >>= 1) {
    if (tid < off) red[tid] += red[tid + off];
    __syncthreads();
  }
  float r = red[0];
  __syncthreads();
  return r;
}

// ---------------- gather: x = item_emb[input_ids] ----------------
__global__ __launch_bounds__(128) void gather_kernel(const float* __restrict__ emb,
                                                     const int* __restrict__ ids,
                                                     float* __restrict__ x) {
  const int row = blockIdx.x, tid = threadIdx.x;
  const float* src = emb + (size_t)ids[row] * DIM;
  float* dst = x + (size_t)row * DIM;
  *(float4*)&dst[tid * 4] = *(const float4*)&src[tid * 4];
}

// ---------------- f32 -> bf16 bulk convert (8 elems/thread) ----------------
__global__ __launch_bounds__(128) void convert_kernel(const float* __restrict__ src,
                                                      __bf16* __restrict__ dst) {
  size_t i = ((size_t)blockIdx.x * 128 + threadIdx.x) * 8;
  float4 a = *(const float4*)&src[i];
  float4 b = *(const float4*)&src[i + 4];
  v4u pv = {pack2(a.x, a.y), pack2(a.z, a.w), pack2(b.x, b.y), pack2(b.z, b.w)};
  *(v4u*)&dst[i] = pv;
}

// ---------- convert item_emb -> bf16 [20032][512] (zero pad rows) ----------
__global__ __launch_bounds__(128) void convert_emb_kernel(
    const float* __restrict__ emb, __bf16* __restrict__ ebf) {
  const int row = blockIdx.x, tid = threadIdx.x;
  const int c = tid * 4;
  float4 v = {0.f, 0.f, 0.f, 0.f};
  if (row < NITEMS) v = *(const float4*)&emb[(size_t)row * DIM + c];
  v2u pv = {pack2(v.x, v.y), pack2(v.z, v.w)};
  *(v2u*)&ebf[(size_t)row * DIM + c] = pv;
}

// -------- transpose+convert weight f32 [K][N] -> bf16 [N][K] --------
__global__ __launch_bounds__(256) void transpose_w_kernel(
    const float* __restrict__ W, __bf16* __restrict__ Wt, int K, int N) {
  __shared__ float tile[32][33];
  const int k0 = blockIdx.y * 32, n0 = blockIdx.x * 32;
  const int tx = threadIdx.x, ty = threadIdx.y;
#pragma unroll
  for (int j = 0; j < 4; ++j)
    tile[ty + j * 8][tx] = W[(size_t)(k0 + ty + j * 8) * N + n0 + tx];
  __syncthreads();
#pragma unroll
  for (int j = 0; j < 4; ++j)
    Wt[(size_t)(n0 + ty + j * 8) * K + k0 + tx] = f2bf(tile[tx][ty + j * 8]);
}

// ---------------- WMMA GEMM, async double-buffered tiles ----------------
// 128x64 workgroup tile, 128 threads = 4 waves; wave w owns rows
// [w*32, w*32+32) (two 16-row groups) x all 64 cols: 8 wmma per k-tile fed by
// 12 ds_load_b128 (2 A-frags + 4 B-frags, B reused by both row groups).
template <bool SILU, bool HAS_BIAS, bool ACC_ADD, bool OUT_BF16>
__global__ __launch_bounds__(128) void gemm_kernel(
    const __bf16* __restrict__ A, const __bf16* __restrict__ Bt,
    const float* __restrict__ bias, void* __restrict__ Cv, int M, int N, int K,
    int lda, int ldb, int ldc) {
  constexpr int TS = 40;
  __shared__ __bf16 As[2][128 * TS];  // [m][k]
  __shared__ __bf16 Bs[2][64 * TS];   // [n][k]
  const int tid = threadIdx.x, wave = tid >> 5, lane = tid & 31;
  const int ncol = lane & 15, hi = lane >> 4;
  const int n0 = blockIdx.x * 64, m0 = blockIdx.y * 128;

  // A: 4 async chunks/thread (128x32 tile); B: 2 chunks/thread (64x32 tile)
  unsigned gaof[4], laof[4];
#pragma unroll
  for (int q = 0; q < 4; ++q) {
    int id = q * 128 + tid;
    int r = id >> 2, c = (id & 3) * 8;
    gaof[q] = (unsigned)(((m0 + r) * lda + c) * 2);
    laof[q] = (unsigned)(r * TS + c) * 2;
  }
  const int rb0 = tid >> 2, rb1 = (128 + tid) >> 2;
  const int cb0 = (tid & 3) * 8, cb1 = ((128 + tid) & 3) * 8;
  const unsigned gb0 = (unsigned)(((n0 + rb0) * ldb + cb0) * 2);
  const unsigned gb1 = (unsigned)(((n0 + rb1) * ldb + cb1) * 2);
  const unsigned lb0 = (unsigned)(rb0 * TS + cb0) * 2;
  const unsigned lb1 = (unsigned)(rb1 * TS + cb1) * 2;

  auto issue = [&](int kt, int buf) {
    unsigned kb = (unsigned)kt * 64u;  // 32 bf16 = 64 bytes per k-step
    unsigned labase = lds_addr(&As[buf][0]);
#pragma unroll
    for (int q = 0; q < 4; ++q) async_b128(labase + laof[q], A, gaof[q] + kb);
    unsigned lbbase = lds_addr(&Bs[buf][0]);
    async_b128(lbbase + lb0, Bt, gb0 + kb);
    async_b128(lbbase + lb1, Bt, gb1 + kb);
  };

  v8f acc0[4], acc1[4];
#pragma unroll
  for (int s = 0; s < 4; ++s) {
    acc0[s] = (v8f){0, 0, 0, 0, 0, 0, 0, 0};
    acc1[s] = (v8f){0, 0, 0, 0, 0, 0, 0, 0};
  }

  const int nt = K >> 5;
  issue(0, 0);
  for (int kt = 0; kt < nt; ++kt) {
    const int cur = kt & 1;
    if (kt + 1 < nt) {
      issue(kt + 1, cur ^ 1);
      wait_async6();          // tile kt complete (in-order per wave)
    } else {
      wait_async0();
    }
    __syncthreads();
    v16bf a0 = load_a_frag(&As[cur][(wave * 32) * TS], TS, lane);
    v16bf a1 = load_a_frag(&As[cur][(wave * 32 + 16) * TS], TS, lane);
#pragma unroll
    for (int s = 0; s < 4; ++s) {
      v16bf b = load_b_frag(&Bs[cur][0], TS, s * 16, lane);
      acc0[s] = wmma_bf16(a0, b, acc0[s]);
      acc1[s] = wmma_bf16(a1, b, acc1[s]);
    }
    __syncthreads();
  }

  float* Cf = (float*)Cv;
  __bf16* Ch = (__bf16*)Cv;
#pragma unroll
  for (int g = 0; g < 2; ++g) {
#pragma unroll
    for (int s = 0; s < 4; ++s) {
      int n = n0 + s * 16 + ncol;
      if (n < N) {
        float bv = HAS_BIAS ? bias[n] : 0.0f;
#pragma unroll
        for (int r = 0; r < 8; ++r) {
          int m = m0 + wave * 32 + g * 16 + hi * 8 + r;
          float v = (g ? acc1[s][r] : acc0[s][r]) + bv;
          if (SILU) v = silu(v);
          if (OUT_BF16) {
            Ch[(size_t)m * ldc + n] = f2bf(v);
          } else {
            float* cp = &Cf[(size_t)m * ldc + n];
            if (ACC_ADD) v += *cp;
            *cp = v;
          }
        }
      }
    }
  }
  (void)M;
}

// ---------------- fused SiLU attention with pos/time biases ----------------
__global__ __launch_bounds__(128) void attn_kernel(
    const __bf16* __restrict__ pb_g, const int* __restrict__ ids,
    const int* __restrict__ ts, const float* __restrict__ pos_e,
    const float* __restrict__ time_e, float* __restrict__ out) {
  constexpr int QS = 72, KS = 72, VS = 40, SS = 40;
  __shared__ __bf16 Qs[64 * QS];      // [qrow][hd]
  __shared__ __bf16 Ks[32 * KS];      // [key][hd]
  __shared__ __bf16 Vs[64 * VS];      // [hd][key]
  __shared__ __bf16 Sls[4][16 * SS];  // per-wave SiLU(scores)
  __shared__ int tsq[64];
  __shared__ int tsk[32];
  __shared__ int kpad[32];
  __shared__ float pos_l[32];
  __shared__ float time_l[64];

  const int q0 = blockIdx.x * 64;
  const int h  = blockIdx.y;
  const int b  = blockIdx.z;
  const int tid = threadIdx.x, wave = tid >> 5, lane = tid & 31;
  const int ncol = lane & 15, hi = lane >> 4;
  const __bf16* pbb = pb_g + (size_t)b * LQ * D4;

#pragma unroll
  for (int j = 0; j < 4; ++j) {
    int id = j * 128 + tid;
    int r = id >> 3, c = (id & 7) * 8;
    unsigned ga = (unsigned)(((q0 + r) * D4 + 1024 + h * HD + c) * 2);
    async_b128(lds_addr(&Qs[r * QS + c]), pbb, ga);
  }
  if (tid < 64) tsq[tid] = ts[b * LQ + q0 + tid];
  if (tid < 32) pos_l[tid] = pos_e[tid * NH + h];
  else if (tid < 96) time_l[tid - 32] = time_e[(tid - 32) * NH + h];

  v8f o[4];
#pragma unroll
  for (int s = 0; s < 4; ++s) o[s] = (v8f){0, 0, 0, 0, 0, 0, 0, 0};
  wait_async0();
  __syncthreads();

  const v16bf qa0 = load_a_frag(&Qs[wave * 16 * QS], QS, lane);
  const v16bf qa1 = load_a_frag(&Qs[wave * 16 * QS + 32], QS, lane);

  const int kend = q0 + 64;  // causal
  for (int kt0 = 0; kt0 < kend; kt0 += 32) {
    __syncthreads();
#pragma unroll
    for (int j = 0; j < 2; ++j) {
      int id = j * 128 + tid;
      int r = id >> 3, c = (id & 7) * 8;
      unsigned ga = (unsigned)(((kt0 + r) * D4 + 1536 + h * HD + c) * 2);
      async_b128(lds_addr(&Ks[r * KS + c]), pbb, ga);
    }
    {
      int kp = tid & 15;
      int hc = (tid >> 4) * 8;
      const __bf16* v0p = pbb + (size_t)(kt0 + 2 * kp) * D4 + 512 + h * HD + hc;
      v4u a0 = *(const v4u*)v0p;
      v4u a1 = *(const v4u*)(v0p + D4);
#pragma unroll
      for (int d = 0; d < 4; ++d) {
        unsigned lo = __builtin_amdgcn_perm(a1[d], a0[d], 0x05040100);
        unsigned hh = __builtin_amdgcn_perm(a1[d], a0[d], 0x07060302);
        *(unsigned*)&Vs[(hc + 2 * d) * VS + 2 * kp] = lo;
        *(unsigned*)&Vs[(hc + 2 * d + 1) * VS + 2 * kp] = hh;
      }
    }
    if (tid < 32) {
      tsk[tid] = ts[b * LQ + kt0 + tid];
      kpad[tid] = (ids[b * LQ + kt0 + tid] == 0) ? 1 : 0;
    }
    wait_async0();
    __syncthreads();

    // S = Q @ K^T
    v16bf kb0 = load_b_frag(Ks, KS, 0, lane);
    v16bf kb1 = load_b_frag(Ks + 32, KS, 0, lane);
    v16bf kb2 = load_b_frag(Ks, KS, 16, lane);
    v16bf kb3 = load_b_frag(Ks + 32, KS, 16, lane);
    v8f sc[2];
    sc[0] = (v8f){0, 0, 0, 0, 0, 0, 0, 0};
    sc[1] = (v8f){0, 0, 0, 0, 0, 0, 0, 0};
    sc[0] = wmma_bf16(qa0, kb0, sc[0]);
    sc[0] = wmma_bf16(qa1, kb1, sc[0]);
    sc[1] = wmma_bf16(qa0, kb2, sc[1]);
    sc[1] = wmma_bf16(qa1, kb3, sc[1]);

    // bias + mask + SiLU -> per-wave LDS S tile
#pragma unroll
    for (int sub = 0; sub < 2; ++sub) {
      int kl = sub * 16 + ncol;
      int key = kt0 + kl;
#pragma unroll
      for (int r = 0; r < 8; ++r) {
        int ql = wave * 16 + hi * 8 + r;
        int q = q0 + ql;
        float v = 0.0f;
        if (key <= q && !kpad[kl]) {
          int rp = q - key;
          int pbk;
          if (rp < 16) pbk = rp;
          else {
            pbk = 16 + (int)(__logf((float)rp * 0.0625f) * (16.0f / 2.0794415f));
            pbk = pbk > 31 ? 31 : pbk;
          }
          int td = tsq[ql] - tsk[kl];
          int ad = td < 0 ? -td : td;
          if (ad < 1) ad = 1;
          int tb = (int)(__logf((float)ad) * (1.0f / 0.693f));
          tb = tb > 63 ? 63 : tb;
          v = silu(sc[sub][r] + pos_l[pbk] + time_l[tb]);
        }
        Sls[wave][(hi * 8 + r) * SS + kl] = f2bf(v);
      }
    }

    // out += SiLU(S) @ V
    v16bf sa = load_a_frag(&Sls[wave][0], SS, lane);
    v16bf vb0 = load_b_frag(Vs, VS, 0, lane);
    v16bf vb1 = load_b_frag(Vs, VS, 16, lane);
    v16bf vb2 = load_b_frag(Vs, VS, 32, lane);
    v16bf vb3 = load_b_frag(Vs, VS, 48, lane);
    o[0] = wmma_bf16(sa, vb0, o[0]);
    o[1] = wmma_bf16(sa, vb1, o[1]);
    o[2] = wmma_bf16(sa, vb2, o[2]);
    o[3] = wmma_bf16(sa, vb3, o[3]);
  }

#pragma unroll
  for (int s = 0; s < 4; ++s) {
    int c = s * 16 + ncol;
#pragma unroll
    for (int r = 0; r < 8; ++r) {
      int q = q0 + wave * 16 + hi * 8 + r;
      out[((size_t)(b * LQ + q)) * DIM + h * HD + c] = o[s][r];
    }
  }
}

// ------------- LN -> bf16 out (thread owns 4 contiguous elems) -------------
__global__ __launch_bounds__(128) void ln_bf_kernel(__bf16* __restrict__ yb,
                                                    const float* __restrict__ x,
                                                    const float* __restrict__ w,
                                                    const float* __restrict__ bb) {
  __shared__ float red[128];
  const int row = blockIdx.x, tid = threadIdx.x;
  float4 v = *(const float4*)&x[(size_t)row * DIM + tid * 4];
  float mean = block_reduce(v.x + v.y + v.z + v.w, red, tid) * (1.0f / DIM);
  float dx = v.x - mean, dy = v.y - mean, dz = v.z - mean, dw = v.w - mean;
  float var = block_reduce(dx * dx + dy * dy + dz * dz + dw * dw, red, tid) *
              (1.0f / DIM);
  float rstd = rsqrtf(var + 1e-5f);
  float4 wv = *(const float4*)&w[tid * 4];
  float4 bv = *(const float4*)&bb[tid * 4];
  v2u pv = {pack2(dx * rstd * wv.x + bv.x, dy * rstd * wv.y + bv.y),
            pack2(dz * rstd * wv.z + bv.z, dw * rstd * wv.w + bv.w)};
  *(v2u*)&yb[(size_t)row * DIM + tid * 4] = pv;
}

// x = x + LN(attn)*U   (U = pb[:, 0:512] bf16, row stride 2048)
__global__ __launch_bounds__(128) void ln_gate_add_kernel(
    float* __restrict__ x, const float* __restrict__ attn,
    const __bf16* __restrict__ pb, const float* __restrict__ w,
    const float* __restrict__ bb) {
  __shared__ float red[128];
  const int row = blockIdx.x, tid = threadIdx.x;
  float4 v = *(const float4*)&attn[(size_t)row * DIM + tid * 4];
  float mean = block_reduce(v.x + v.y + v.z + v.w, red, tid) * (1.0f / DIM);
  float dx = v.x - mean, dy = v.y - mean, dz = v.z - mean, dw = v.w - mean;
  float var = block_reduce(dx * dx + dy * dy + dz * dz + dw * dw, red, tid) *
              (1.0f / DIM);
  float rstd = rsqrtf(var + 1e-5f);
  float4 wv = *(const float4*)&w[tid * 4];
  float4 bv = *(const float4*)&bb[tid * 4];
  v4bf ub = *(const v4bf*)&pb[(size_t)row * D4 + tid * 4];
  float4 xv = *(float4*)&x[(size_t)row * DIM + tid * 4];
  xv.x += (dx * rstd * wv.x + bv.x) * bf2f(ub[0]);
  xv.y += (dy * rstd * wv.y + bv.y) * bf2f(ub[1]);
  xv.z += (dz * rstd * wv.z + bv.z) * bf2f(ub[2]);
  xv.w += (dw * rstd * wv.w + bv.w) * bf2f(ub[3]);
  *(float4*)&x[(size_t)row * DIM + tid * 4] = xv;
}

// ---------------- launch ----------------
extern "C" void kernel_launch(void* const* d_in, const int* in_sizes, int n_in,
                              void* d_out, int out_size, void* d_ws,
                              size_t ws_size, hipStream_t stream) {
  const int*   input_ids  = (const int*)d_in[0];
  const int*   timestamps = (const int*)d_in[1];
  const float* item_emb   = (const float*)d_in[2];
  const float* proj_w     = (const float*)d_in[3];
  const float* proj_b     = (const float*)d_in[4];
  const float* pos_emb    = (const float*)d_in[5];
  const float* time_emb   = (const float*)d_in[6];
  const float* an_w       = (const float*)d_in[7];
  const float* an_b       = (const float*)d_in[8];
  const float* f1_w       = (const float*)d_in[9];
  const float* f1_b       = (const float*)d_in[10];
  const float* f2_w       = (const float*)d_in[11];
  const float* f2_b       = (const float*)d_in[12];
  const float* fn_w       = (const float*)d_in[13];
  const float* fn_b       = (const float*)d_in[14];
  const float* final_w    = (const float*)d_in[15];
  const float* final_b    = (const float*)d_in[16];
  float* out = (float*)d_out;

  size_t off = 0;
  auto walloc = [&](size_t bytes) {
    void* ptr = (char*)d_ws + off;
    off += (bytes + 255) & ~(size_t)255;
    return ptr;
  };
  float*  x    = (float*)walloc((size_t)BL * DIM * 4);
  float*  attn = (float*)walloc((size_t)BL * DIM * 4);
  __bf16* xb   = (__bf16*)walloc((size_t)BL * DIM * 2);
  __bf16* yb   = (__bf16*)walloc((size_t)BL * DIM * 2);
  __bf16* pb   = (__bf16*)walloc((size_t)BL * D4 * 2);
  __bf16* hb   = (__bf16*)walloc((size_t)BL * D4 * 2);
  __bf16* pwt  = (__bf16*)walloc((size_t)2 * D4 * DIM * 2);
  __bf16* f1t  = (__bf16*)walloc((size_t)2 * D4 * DIM * 2);
  __bf16* f2t  = (__bf16*)walloc((size_t)2 * DIM * D4 * 2);
  __bf16* ebf  = (__bf16*)walloc((size_t)NPAD * DIM * 2);

  gather_kernel<<<BL, 128, 0, stream>>>(item_emb, input_ids, x);
  convert_emb_kernel<<<NPAD, 128, 0, stream>>>(item_emb, ebf);
  for (int i = 0; i < 2; ++i) {
    transpose_w_kernel<<<dim3(D4 / 32, DIM / 32), dim3(32, 8), 0, stream>>>(
        proj_w + (size_t)i * DIM * D4, pwt + (size_t)i * D4 * DIM, DIM, D4);
    transpose_w_kernel<<<dim3(D4 / 32, DIM / 32), dim3(32, 8), 0, stream>>>(
        f1_w + (size_t)i * DIM * D4, f1t + (size_t)i * D4 * DIM, DIM, D4);
    transpose_w_kernel<<<dim3(DIM / 32, D4 / 32), dim3(32, 8), 0, stream>>>(
        f2_w + (size_t)i * D4 * DIM, f2t + (size_t)i * DIM * D4, D4, DIM);
  }

  for (int i = 0; i < 2; ++i) {
    convert_kernel<<<(BL * DIM) / 1024, 128, 0, stream>>>(x, xb);
    gemm_kernel<true, true, false, true><<<dim3(D4 / 64, BL / 128), 128, 0, stream>>>(
        xb, pwt + (size_t)i * D4 * DIM, proj_b + (size_t)i * D4, pb,
        BL, D4, DIM, DIM, DIM, D4);
    attn_kernel<<<dim3(LQ / 64, NH, BQ), 128, 0, stream>>>(
        pb, input_ids, timestamps, pos_emb + (size_t)i * 32 * NH,
        time_emb + (size_t)i * 64 * NH, attn);
    ln_gate_add_kernel<<<BL, 128, 0, stream>>>(x, attn, pb,
                                               an_w + (size_t)i * DIM,
                                               an_b + (size_t)i * DIM);
    ln_bf_kernel<<<BL, 128, 0, stream>>>(yb, x, fn_w + (size_t)i * DIM,
                                         fn_b + (size_t)i * DIM);
    gemm_kernel<true, true, false, true><<<dim3(D4 / 64, BL / 128), 128, 0, stream>>>(
        yb, f1t + (size_t)i * D4 * DIM, f1_b + (size_t)i * D4, hb,
        BL, D4, DIM, DIM, DIM, D4);
    gemm_kernel<false, true, true, false><<<dim3(DIM / 64, BL / 128), 128, 0, stream>>>(
        hb, f2t + (size_t)i * DIM * D4, f2_b + (size_t)i * DIM, x,
        BL, DIM, D4, D4, D4, DIM);
  }

  ln_bf_kernel<<<BL, 128, 0, stream>>>(yb, x, final_w, final_b);
  gemm_kernel<false, false, false, false><<<dim3(NPAD / 64, BL / 128), 128, 0, stream>>>(
      yb, ebf, nullptr, out, BL, NITEMS, DIM, DIM, DIM, NITEMS);

  (void)in_sizes; (void)n_in; (void)out_size; (void)ws_size;
}
